// TransformerBlock_32040456028778
// MI455X (gfx1250) — compile-verified, matrix operand order checked by reading, not run on hardware
//
#include <hip/hip_runtime.h>

typedef __attribute__((ext_vector_type(16))) __bf16 bf16x16;
typedef __attribute__((ext_vector_type(8)))  float  f32x8;
typedef __attribute__((ext_vector_type(4))) unsigned int u32x4;
typedef __attribute__((ext_vector_type(8))) int i32x8;
typedef __attribute__((ext_vector_type(4))) int i32x4;
typedef unsigned short u16;
typedef unsigned int   u32;
typedef unsigned long long u64;

#define DIMSZ 1024
#define SEQ   4096
#define NH    16
#define HDIM  64
#define HID   2816

// ---------------- bf16 helpers ----------------
__device__ __forceinline__ u16 f2bf(float f) {
  u32 u = __float_as_uint(f);
  u32 r = u + 0x7FFFu + ((u >> 16) & 1u);   // round-to-nearest-even
  return (u16)(r >> 16);
}
__device__ __forceinline__ float bf2f(u16 b) {
  return __uint_as_float(((u32)b) << 16);
}

union FragU { u32 u[8]; bf16x16 v; };

// Load a 16x32 bf16 WMMA A/B fragment from LDS (row-major [row][k], ld in u16, even).
// Lane m = lane&15, half = lane>>4; dwords: p[half*4+j] (K=half*8+2j), p[8+half*4+j] (K=16+...).
__device__ __forceinline__ bf16x16 load_frag(const u16* s, int ld, int row0, int k0) {
  int lane = threadIdx.x & 31;
  int half = lane >> 4;
  const u32* p = (const u32*)(s + (size_t)(row0 + (lane & 15)) * ld + k0);
  FragU f;
#pragma unroll
  for (int j = 0; j < 4; ++j) f.u[j] = p[half * 4 + j];
#pragma unroll
  for (int j = 0; j < 4; ++j) f.u[4 + j] = p[8 + half * 4 + j];
  return f.v;
}

__device__ __forceinline__ f32x8 wmma_bf16(bf16x16 a, bf16x16 b, f32x8 c) {
  return __builtin_amdgcn_wmma_f32_16x16x32_bf16(false, a, false, b, (short)0, c, false, false);
}

// ---------------- CDNA5 async memory->LDS copy (ASYNCcnt tracked) ----------------
// LDS operand is the low 32 bits of the generic pointer (ISA 10.2: LDS addr = addr[31:0]).
__device__ __forceinline__ u32 lds_addr32(const void* p) {
  return (u32)(uintptr_t)p;
}
__device__ __forceinline__ void async_copy_b128(void* lds_dst, const void* gsrc) {
  u32 l = lds_addr32(lds_dst);
  asm volatile("global_load_async_to_lds_b128 %0, %1, off"
               :: "v"(l), "v"(gsrc) : "memory");
}
#define WAIT_ASYNCCNT(n) asm volatile("s_wait_asynccnt " #n ::: "memory")

// ---------------- CDNA5 Tensor Data Mover: 2D tile (128 rows x 64 bf16) -> LDS ------------
// D# per CDNA5 ISA. Row stride in memory = row_stride_elems (bf16).
// LDS rows are padded by 16B (pad_interval code 4 = 32 DWORDs = 128B row; pad_amount code 3
// = 4 DWORDs = 16B), reproducing the LD=72 u16 LDS pitch. tensor dims == tile dims => no OOB.
__device__ __forceinline__ void tdm_load_tile_128x64(const u16* gsrc, u64 row_stride_elems,
                                                     void* lds_dst) {
  u64 ga = (u64)(uintptr_t)gsrc;
  u32x4 g0;
  g0[0] = 1u;                                           // count=1, user descriptor
  g0[1] = lds_addr32(lds_dst);                          // lds_addr
  g0[2] = (u32)(ga & 0xFFFFFFFFu);                      // global_addr[31:0]
  g0[3] = (u32)((ga >> 32) & 0x1FFFFFFu) | 0x80000000u; // global_addr[56:32] | type=2
  i32x8 g1;
  g1[0] = (1 << 16)        // data_size = 1 (2 bytes)
        | (1 << 20)        // pad_enable
        | (4 << 22)        // pad_interval: 32 DWORDs (128B) between pads
        | (3 << 25);       // pad_amount: 4 DWORDs (16B)
  g1[1] = (int)(64u << 16);            // tensor_dim0[15:0] = 64 (bits 63:48 of group)
  g1[2] = (int)(128u << 16);           // tensor_dim0[31:16]=0 ; tensor_dim1[15:0] = 128
  g1[3] = (int)(64u << 16);            // tensor_dim1[31:16]=0 ; tile_dim0 = 64
  g1[4] = (int)128;                    // tile_dim1 = 128 ; tile_dim2 = 0
  g1[5] = (int)(u32)(row_stride_elems & 0xFFFFFFFFu);   // tensor_dim0_stride[31:0]
  g1[6] = (int)(u32)((row_stride_elems >> 32) & 0xFFFFu); // stride[47:32]; dim1_stride lo = 0
  g1[7] = 0;
  i32x4 z4 = {0, 0, 0, 0};
#if defined(__clang_major__) && __clang_major__ >= 23
  i32x8 z8 = {0, 0, 0, 0, 0, 0, 0, 0};
  __builtin_amdgcn_tensor_load_to_lds(g0, g1, z4, z4, z8, 0);
#else
  __builtin_amdgcn_tensor_load_to_lds(g0, g1, z4, z4, 0);
#endif
}

// ---------------- weight transpose + fp32->bf16 ----------------
// w is [K][N] fp32 row-major; wt is [N][K] bf16 row-major (B^T for the GEMM).
__global__ void transpose_to_bf16(const float* __restrict__ w, u16* __restrict__ wt,
                                  int K, int N) {
  size_t i = (size_t)blockIdx.x * blockDim.x + threadIdx.x;
  size_t tot = (size_t)K * N;
  if (i >= tot) return;
  size_t n = i / K, k = i - n * K;
  wt[i] = f2bf(w[k * (size_t)N + n]);
}

// ---------------- RMSNorm (fp32 in, bf16 out) ----------------
__global__ __launch_bounds__(256) void rmsnorm_bf16(const float* __restrict__ x,
                                                    const float* __restrict__ g,
                                                    u16* __restrict__ o) {
  int row = blockIdx.x;
  int tid = threadIdx.x;
  const float* xr = x + (size_t)row * DIMSZ;
  float v[4];
  float ss = 0.f;
#pragma unroll
  for (int i = 0; i < 4; ++i) { v[i] = xr[tid + i * 256]; ss += v[i] * v[i]; }
#pragma unroll
  for (int off = 16; off; off >>= 1) ss += __shfl_xor(ss, off);
  __shared__ float red[8];
  if ((tid & 31) == 0) red[tid >> 5] = ss;
  __syncthreads();
  float tot = 0.f;
#pragma unroll
  for (int i = 0; i < 8; ++i) tot += red[i];
  float rinv = rsqrtf(tot * (1.0f / DIMSZ) + 1e-5f);
#pragma unroll
  for (int i = 0; i < 4; ++i) {
    int d = tid + i * 256;
    o[(size_t)row * DIMSZ + d] = f2bf(v[i] * rinv * g[d]);
  }
}

// ---------------- tiled WMMA GEMM: C[M][N] = A[M][K] * Bt[N][K]^T (+resid) ----------------
// 256 threads = 8 waves; block tile 128x128, K-block 64, double-buffered TDM staging.
// Wave 0 issues one TENSOR_LOAD_TO_LDS per matrix per K-block (TENSORcnt tracked);
// waves: 4 (M) x 2 (N); per wave 32 rows x 64 cols = 2x4 WMMA tiles; 16 wmma per K-block.
__global__ __launch_bounds__(256) void gemm_bf16_bt(
    const u16* __restrict__ A, const u16* __restrict__ Bt,
    int M, int N, int K, const float* __restrict__ resid,
    void* __restrict__ out, int out_is_bf16) {
  const int LD = 72;  // 64 + 8 pad (u16); row = 144B (TDM pad reproduces this pitch)
  __shared__ u16 As[2][128 * 72];
  __shared__ u16 Bs[2][128 * 72];
  int bm = blockIdx.y * 128, bn = blockIdx.x * 128;
  int tid = threadIdx.x;
  int wave = tid >> 5, lane = tid & 31;
  int rowA0 = (wave >> 1) * 32;
  int colB0 = (wave & 1) * 64;

  f32x8 acc[2][4];
#pragma unroll
  for (int i = 0; i < 2; ++i)
#pragma unroll
    for (int j = 0; j < 4; ++j)
#pragma unroll
      for (int r = 0; r < 8; ++r) acc[i][j][r] = 0.f;

  // wave 0 issues the TDM tile loads for one K-block into LDS buffer `buf`
  auto issue_block = [&](int buf, int k0) {
    tdm_load_tile_128x64(A + (size_t)bm * K + k0, (u64)K, &As[buf][0]);
    tdm_load_tile_128x64(Bt + (size_t)bn * K + k0, (u64)K, &Bs[buf][0]);
  };

  int nblk = K >> 6;
  if (wave == 0) issue_block(0, 0);
  for (int i = 0; i < nblk; ++i) {
    int cur = i & 1;
    if (wave == 0) {
      if (i + 1 < nblk) {
        issue_block(cur ^ 1, (i + 1) << 6);
        __builtin_amdgcn_s_wait_tensorcnt(2);  // in-order: current buffer's 2 loads done
      } else {
        __builtin_amdgcn_s_wait_tensorcnt(0);
      }
    }
    __syncthreads();   // publishes TDM-written LDS to all waves

    const u16* as = As[cur];
    const u16* bs = Bs[cur];
#pragma unroll
    for (int ks = 0; ks < 64; ks += 32) {
      bf16x16 a0 = load_frag(as, LD, rowA0, ks);
      bf16x16 a1 = load_frag(as, LD, rowA0 + 16, ks);
      bf16x16 b0 = load_frag(bs, LD, colB0, ks);
      bf16x16 b1 = load_frag(bs, LD, colB0 + 16, ks);
      bf16x16 b2 = load_frag(bs, LD, colB0 + 32, ks);
      bf16x16 b3 = load_frag(bs, LD, colB0 + 48, ks);
      acc[0][0] = wmma_bf16(a0, b0, acc[0][0]);
      acc[0][1] = wmma_bf16(a0, b1, acc[0][1]);
      acc[0][2] = wmma_bf16(a0, b2, acc[0][2]);
      acc[0][3] = wmma_bf16(a0, b3, acc[0][3]);
      acc[1][0] = wmma_bf16(a1, b0, acc[1][0]);
      acc[1][1] = wmma_bf16(a1, b1, acc[1][1]);
      acc[1][2] = wmma_bf16(a1, b2, acc[1][2]);
      acc[1][3] = wmma_bf16(a1, b3, acc[1][3]);
    }
    __syncthreads();   // all reads of buf `cur` done before it is overwritten next round
  }

  int mb = bm + rowA0, nb = bn + colB0;
#pragma unroll
  for (int i = 0; i < 2; ++i)
#pragma unroll
    for (int jt = 0; jt < 4; ++jt)
#pragma unroll
      for (int r = 0; r < 8; ++r) {
        int mm = mb + i * 16 + (lane >> 4) * 8 + r;
        int nn = nb + jt * 16 + (lane & 15);
        float vv = acc[i][jt][r];
        if (resid) vv += resid[(size_t)mm * N + nn];
        if (out_is_bf16) ((u16*)out)[(size_t)mm * N + nn] = f2bf(vv);
        else             ((float*)out)[(size_t)mm * N + nn] = vv;
      }
}

// ---------------- V transpose: v[s][h*64+d] -> vt[(h*64+d)][s] (bf16) ----------------
__global__ void v_transpose_bf16(const u16* __restrict__ v, u16* __restrict__ vt) {
  size_t i = (size_t)blockIdx.x * blockDim.x + threadIdx.x;
  if (i >= (size_t)SEQ * DIMSZ) return;
  size_t hd = i >> 12;        // / SEQ
  size_t s  = i & (SEQ - 1);
  vt[i] = v[s * DIMSZ + hd];
}

// ---------------- flash attention ----------------
// grid (SEQ/64, NH), 128 threads = 4 waves, each wave owns 16 query rows.
// Q,K bf16 [SEQ][DIMSZ] (col = h*64+d); Vt bf16 [(h*64+d)][SEQ]. Out bf16 [SEQ][DIMSZ].
// K/Vt tiles double-buffered via async-to-LDS (ASYNCcnt path).
__global__ __launch_bounds__(128) void flash_attn_bf16(
    const u16* __restrict__ Q, const u16* __restrict__ Kb,
    const u16* __restrict__ Vt, u16* __restrict__ Out) {
  const int LD = 72;  // 64 + 8 pad
  __shared__ u16 Qs[64 * 72];
  __shared__ u16 Ks[2][64 * 72];
  __shared__ u16 Vs[2][64 * 72];     // [d][key]
  __shared__ u16 Ps[4 * 16 * 72];    // per-wave P tile 16x64

  int h = blockIdx.y, qb = blockIdx.x;
  int tid = threadIdx.x, wave = tid >> 5, lane = tid & 31;

  // per-thread staging coords: 512 chunks of 8 bf16 per 64x64 tile -> 4 per thread
  int tr[4], tc[4];
#pragma unroll
  for (int p = 0; p < 4; ++p) {
    int idx = p * 128 + tid;
    tr[p] = idx >> 3;
    tc[p] = (idx & 7) * 8;
  }

  auto issue_kv = [&](int buf, int j) {
#pragma unroll
    for (int p = 0; p < 4; ++p) {
      async_copy_b128(&Ks[buf][tr[p] * LD + tc[p]],
                      Kb + (size_t)(j * 64 + tr[p]) * DIMSZ + h * HDIM + tc[p]);
      async_copy_b128(&Vs[buf][tr[p] * LD + tc[p]],
                      Vt + (size_t)(h * HDIM + tr[p]) * SEQ + j * 64 + tc[p]);
    }
  };

  // stage Q tile (4 asyncs/thread), then K/V block 0 (8 asyncs/thread)
#pragma unroll
  for (int p = 0; p < 4; ++p)
    async_copy_b128(&Qs[tr[p] * LD + tc[p]],
                    Q + (size_t)(qb * 64 + tr[p]) * DIMSZ + h * HDIM + tc[p]);
  issue_kv(0, 0);
  WAIT_ASYNCCNT(0x8);   // Q's 4 issues complete (8 newer K/V issues may remain)
  __syncthreads();

  bf16x16 qf0 = load_frag(Qs, LD, wave * 16, 0);
  bf16x16 qf1 = load_frag(Qs, LD, wave * 16, 32);

  float m[8], l[8];
  f32x8 o[4];
#pragma unroll
  for (int r = 0; r < 8; ++r) { m[r] = -1e30f; l[r] = 0.f; }
#pragma unroll
  for (int dt = 0; dt < 4; ++dt)
#pragma unroll
    for (int r = 0; r < 8; ++r) o[dt][r] = 0.f;

  u16* PsW = Ps + wave * 16 * LD;

  for (int j = 0; j <= qb; ++j) {
    int cur = j & 1;
    if (j < qb) {
      issue_kv(cur ^ 1, j + 1);
      WAIT_ASYNCCNT(0x8);
    } else {
      WAIT_ASYNCCNT(0x0);
    }
    __syncthreads();
    const u16* ks = Ks[cur];
    const u16* vs = Vs[cur];

    // S = Q K^T (per wave: 16 rows x 64 keys), fp32
    f32x8 st[4];
#pragma unroll
    for (int nt = 0; nt < 4; ++nt) {
#pragma unroll
      for (int r = 0; r < 8; ++r) st[nt][r] = 0.f;
      st[nt] = wmma_bf16(qf0, load_frag(ks, LD, nt * 16, 0),  st[nt]);
      st[nt] = wmma_bf16(qf1, load_frag(ks, LD, nt * 16, 32), st[nt]);
    }

    // scale + causal mask
    int qrow0 = qb * 64 + wave * 16 + (lane >> 4) * 8;
    int kcol  = j * 64 + (lane & 15);
#pragma unroll
    for (int nt = 0; nt < 4; ++nt)
#pragma unroll
      for (int r = 0; r < 8; ++r) {
        float v = st[nt][r] * 0.125f;  // 1/sqrt(64)
        if (j == qb && (kcol + nt * 16) > (qrow0 + r)) v = -1e30f;
        st[nt][r] = v;
      }

    // online softmax (row lives across 16 lanes of a half-wave)
    float mnew[8], alpha[8];
#pragma unroll
    for (int r = 0; r < 8; ++r) {
      float v = fmaxf(fmaxf(st[0][r], st[1][r]), fmaxf(st[2][r], st[3][r]));
#pragma unroll
      for (int off = 1; off < 16; off <<= 1) v = fmaxf(v, __shfl_xor(v, off));
      mnew[r] = fmaxf(m[r], v);
      alpha[r] = __expf(m[r] - mnew[r]);
      m[r] = mnew[r];
    }
#pragma unroll
    for (int nt = 0; nt < 4; ++nt)
#pragma unroll
      for (int r = 0; r < 8; ++r) st[nt][r] = __expf(st[nt][r] - mnew[r]);
#pragma unroll
    for (int r = 0; r < 8; ++r) {
      float s = st[0][r] + st[1][r] + st[2][r] + st[3][r];
#pragma unroll
      for (int off = 1; off < 16; off <<= 1) s += __shfl_xor(s, off);
      l[r] = l[r] * alpha[r] + s;
    }
#pragma unroll
    for (int dt = 0; dt < 4; ++dt)
#pragma unroll
      for (int r = 0; r < 8; ++r) o[dt][r] *= alpha[r];

    // repack P (C-layout) -> LDS -> A-fragment layout (same-wave DS ops stay ordered)
#pragma unroll
    for (int nt = 0; nt < 4; ++nt)
#pragma unroll
      for (int r = 0; r < 8; ++r)
        PsW[((lane >> 4) * 8 + r) * LD + nt * 16 + (lane & 15)] = f2bf(st[nt][r]);

    bf16x16 p0 = load_frag(PsW, LD, 0, 0);
    bf16x16 p1 = load_frag(PsW, LD, 0, 32);
#pragma unroll
    for (int dt = 0; dt < 4; ++dt) {
      o[dt] = wmma_bf16(p0, load_frag(vs, LD, dt * 16, 0),  o[dt]);
      o[dt] = wmma_bf16(p1, load_frag(vs, LD, dt * 16, 32), o[dt]);
    }
    __syncthreads();
  }

  // normalize + store bf16 to [SEQ][DIMSZ]
#pragma unroll
  for (int r = 0; r < 8; ++r) {
    float inv = 1.f / l[r];
    int qrow = qb * 64 + wave * 16 + (lane >> 4) * 8 + r;
#pragma unroll
    for (int dt = 0; dt < 4; ++dt) {
      int col = h * HDIM + dt * 16 + (lane & 15);
      Out[(size_t)qrow * DIMSZ + col] = f2bf(o[dt][r] * inv);
    }
  }
}

// ---------------- SwiGLU elementwise ----------------
__global__ void silu_mul_bf16(const u16* __restrict__ a, const u16* __restrict__ b,
                              u16* __restrict__ o, size_t n) {
  size_t i = (size_t)blockIdx.x * blockDim.x + threadIdx.x;
  if (i >= n) return;
  float x = bf2f(a[i]);
  float y = bf2f(b[i]);
  float s = x / (1.f + __expf(-x));
  o[i] = f2bf(s * y);
}

// ---------------- host launcher ----------------
extern "C" void kernel_launch(void* const* d_in, const int* in_sizes, int n_in,
                              void* d_out, int out_size, void* d_ws, size_t ws_size,
                              hipStream_t stream) {
  (void)in_sizes; (void)n_in; (void)out_size; (void)ws_size;
  const float* x  = (const float*)d_in[0];
  const float* wq = (const float*)d_in[1];
  const float* wk = (const float*)d_in[2];
  const float* wv = (const float*)d_in[3];
  const float* wo = (const float*)d_in[4];
  const float* w1 = (const float*)d_in[5];
  const float* w2 = (const float*)d_in[6];
  const float* w3 = (const float*)d_in[7];
  const float* ga = (const float*)d_in[8];
  const float* gf = (const float*)d_in[9];
  float* outp = (float*)d_out;

  char* ws = (char*)d_ws;
  size_t off = 0;
  auto alloc = [&](size_t bytes) -> char* {
    char* p = ws + off;
    off += (bytes + 255) & ~(size_t)255;
    return p;
  };

  u16* wqT = (u16*)alloc((size_t)DIMSZ * DIMSZ * 2);
  u16* wkT = (u16*)alloc((size_t)DIMSZ * DIMSZ * 2);
  u16* wvT = (u16*)alloc((size_t)DIMSZ * DIMSZ * 2);
  u16* woT = (u16*)alloc((size_t)DIMSZ * DIMSZ * 2);
  u16* w1T = (u16*)alloc((size_t)DIMSZ * HID * 2);
  u16* w3T = (u16*)alloc((size_t)DIMSZ * HID * 2);
  u16* w2T = (u16*)alloc((size_t)DIMSZ * HID * 2);
  u16* xn  = (u16*)alloc((size_t)SEQ * DIMSZ * 2);
  u16* qb_ = (u16*)alloc((size_t)SEQ * DIMSZ * 2);
  u16* kb_ = (u16*)alloc((size_t)SEQ * DIMSZ * 2);
  u16* vb_ = (u16*)alloc((size_t)SEQ * DIMSZ * 2);
  u16* vtb = (u16*)alloc((size_t)SEQ * DIMSZ * 2);
  u16* atb = (u16*)alloc((size_t)SEQ * DIMSZ * 2);
  float* hres = (float*)alloc((size_t)SEQ * DIMSZ * 4);
  u16* hn  = (u16*)alloc((size_t)SEQ * DIMSZ * 2);
  u16* g1b = (u16*)alloc((size_t)SEQ * HID * 2);
  u16* g3b = (u16*)alloc((size_t)SEQ * HID * 2);
  u16* ffb = (u16*)alloc((size_t)SEQ * HID * 2);

  // 1. weights -> bf16 transposed
  {
    int totD = DIMSZ * DIMSZ;
    int gD = (totD + 255) / 256;
    transpose_to_bf16<<<gD, 256, 0, stream>>>(wq, wqT, DIMSZ, DIMSZ);
    transpose_to_bf16<<<gD, 256, 0, stream>>>(wk, wkT, DIMSZ, DIMSZ);
    transpose_to_bf16<<<gD, 256, 0, stream>>>(wv, wvT, DIMSZ, DIMSZ);
    transpose_to_bf16<<<gD, 256, 0, stream>>>(wo, woT, DIMSZ, DIMSZ);
    int totH = DIMSZ * HID;
    int gH = (totH + 255) / 256;
    transpose_to_bf16<<<gH, 256, 0, stream>>>(w1, w1T, DIMSZ, HID);   // -> [HID][DIM]
    transpose_to_bf16<<<gH, 256, 0, stream>>>(w3, w3T, DIMSZ, HID);   // -> [HID][DIM]
    transpose_to_bf16<<<gH, 256, 0, stream>>>(w2, w2T, HID, DIMSZ);   // -> [DIM][HID]
  }

  // 2. xn = rmsnorm(x) * g_attn  (bf16)
  rmsnorm_bf16<<<SEQ, 256, 0, stream>>>(x, ga, xn);

  // 3. Q,K,V GEMMs (bf16 out)
  {
    dim3 g(DIMSZ / 128, SEQ / 128);
    gemm_bf16_bt<<<g, 256, 0, stream>>>(xn, wqT, SEQ, DIMSZ, DIMSZ, nullptr, qb_, 1);
    gemm_bf16_bt<<<g, 256, 0, stream>>>(xn, wkT, SEQ, DIMSZ, DIMSZ, nullptr, kb_, 1);
    gemm_bf16_bt<<<g, 256, 0, stream>>>(xn, wvT, SEQ, DIMSZ, DIMSZ, nullptr, vb_, 1);
  }

  // 4. V -> V^T per head
  {
    size_t tot = (size_t)SEQ * DIMSZ;
    v_transpose_bf16<<<(unsigned)((tot + 255) / 256), 256, 0, stream>>>(vb_, vtb);
  }

  // 5. flash attention
  {
    dim3 g(SEQ / 64, NH);
    flash_attn_bf16<<<g, 128, 0, stream>>>(qb_, kb_, vtb, atb);
  }

  // 6. h = attn @ wo + x (fp32)
  {
    dim3 g(DIMSZ / 128, SEQ / 128);
    gemm_bf16_bt<<<g, 256, 0, stream>>>(atb, woT, SEQ, DIMSZ, DIMSZ, x, hres, 0);
  }

  // 7. hn = rmsnorm(h) * g_ffn (bf16)
  rmsnorm_bf16<<<SEQ, 256, 0, stream>>>(hres, gf, hn);

  // 8. g1 = hn @ w1, g3 = hn @ w3 (bf16)
  {
    dim3 g(HID / 128, SEQ / 128);
    gemm_bf16_bt<<<g, 256, 0, stream>>>(hn, w1T, SEQ, HID, DIMSZ, nullptr, g1b, 1);
    gemm_bf16_bt<<<g, 256, 0, stream>>>(hn, w3T, SEQ, HID, DIMSZ, nullptr, g3b, 1);
  }

  // 9. ff = silu(g1) * g3 (bf16)
  {
    size_t tot = (size_t)SEQ * HID;
    silu_mul_bf16<<<(unsigned)((tot + 255) / 256), 256, 0, stream>>>(g1b, g3b, ffb, tot);
  }

  // 10. out = ff @ w2 + h (fp32 -> d_out)
  {
    dim3 g(DIMSZ / 128, SEQ / 128);
    gemm_bf16_bt<<<g, 256, 0, stream>>>(ffb, w2T, SEQ, DIMSZ, HID, hres, outp, 0);
  }
}
